// QuantizedLinear_85272280695338
// MI455X (gfx1250) — matrix operand order repaired and checked by reading, not compiled, not run on hardware
//
#include <hip/hip_runtime.h>

typedef __attribute__((ext_vector_type(16))) _Float16 v16h;
typedef __attribute__((ext_vector_type(8)))  _Float16 v8h;
typedef __attribute__((ext_vector_type(8)))  float    v8f;

#define TILE_M 256
#define TILE_N 256
#define TILE_K 32
#define NTHREADS 512
#define LDS_STRIDE 40   // 32 f16 + 8 pad -> 80 bytes/row, 16B-aligned, conflict-free for 16-row groups

__global__ __launch_bounds__(NTHREADS)
void qlinear_wmma_f16(const float* __restrict__ x,     // [M, K] fp32 activations
                      const int*   __restrict__ w,     // [N, K] int8 values stored as int32
                      const float* __restrict__ scale, // [N]
                      const float* __restrict__ bias,  // [N]
                      float* __restrict__ out,         // [M, N]
                      int M, int K, int N)
{
    __shared__ _Float16 As[2][TILE_M * LDS_STRIDE];   // 2 x 20 KB
    __shared__ _Float16 Bs[2][TILE_N * LDS_STRIDE];   // 2 x 20 KB  (80 KB total)

    const int tid  = threadIdx.x;
    const int lane = tid & 31;
    const int wave = tid >> 5;       // 0..15
    const int wm   = wave >> 2;      // 0..3 : 64-row M strip
    const int wn   = wave & 3;       // 0..3 : 64-col N strip
    const int half = lane >> 4;      // 0/1  : lane half
    const int l16  = lane & 15;

    const int mBase = blockIdx.x * TILE_M;
    const int nBase = blockIdx.y * TILE_N;

    // staging: each thread loads 4 x float4 (A) and 4 x int4 (B) per K-step
    const int rs = tid >> 3;         // 0..63 row within a 64-row group
    const int cs = tid & 7;          // 0..7  float4 column within the 32-wide K slab

    v8f acc[4][4] = {};              // 4 (M tiles) x 4 (N tiles)

    float4 ax[4];
    int4   bw[4];

    const int nk = K / TILE_K;

    // ---- prologue: stage K-step 0 into buffer 0 ----
    #pragma unroll
    for (int i = 0; i < 4; ++i) {
        const int r = i * 64 + rs;
        ax[i] = *(const float4*)(x + (size_t)(mBase + r) * K + cs * 4);
        bw[i] = *(const int4*)  (w + (size_t)(nBase + r) * K + cs * 4);
    }
    #pragma unroll
    for (int i = 0; i < 4; ++i) {
        const int r = i * 64 + rs;
        _Float16* pa = &As[0][r * LDS_STRIDE + cs * 4];
        pa[0] = (_Float16)ax[i].x; pa[1] = (_Float16)ax[i].y;
        pa[2] = (_Float16)ax[i].z; pa[3] = (_Float16)ax[i].w;
        _Float16* pb = &Bs[0][r * LDS_STRIDE + cs * 4];
        pb[0] = (_Float16)bw[i].x; pb[1] = (_Float16)bw[i].y;
        pb[2] = (_Float16)bw[i].z; pb[3] = (_Float16)bw[i].w;
    }
    __syncthreads();

    // ---- main K loop: register-staged double buffer, one barrier per step ----
    for (int kt = 0; kt < nk; ++kt) {
        const int cur = kt & 1;
        const int nxt = cur ^ 1;

        if (kt + 1 < nk) {
            const int k0 = (kt + 1) * TILE_K;
            #pragma unroll
            for (int i = 0; i < 4; ++i) {
                const int r = i * 64 + rs;
                ax[i] = *(const float4*)(x + (size_t)(mBase + r) * K + k0 + cs * 4);
                bw[i] = *(const int4*)  (w + (size_t)(nBase + r) * K + k0 + cs * 4);
            }
            if (kt + 2 < nk) {
                const int kp = (kt + 2) * TILE_K;
                __builtin_prefetch(x + (size_t)(mBase + rs) * K + kp, 0, 1);
                __builtin_prefetch(w + (size_t)(nBase + rs) * K + kp, 0, 1);
            }
        }

        // B fragments (32x16 f16): lane%16 = N col, lane/16 = K half; b[j] = B[half*16+j][n]
        v16h bfrag[4];
        #pragma unroll
        for (int u = 0; u < 4; ++u) {
            const _Float16* br = &Bs[cur][(wn * 64 + u * 16 + l16) * LDS_STRIDE + half * 16];
            v8h lo = *(const v8h*)(br);
            v8h hi = *(const v8h*)(br + 8);
            bfrag[u] = __builtin_shufflevector(lo, hi,
                        0,1,2,3,4,5,6,7,8,9,10,11,12,13,14,15);
        }

        // A fragments (16x32 f16): a[j] = A[m][half*16+j]
        #pragma unroll
        for (int t = 0; t < 4; ++t) {
            const _Float16* ar = &As[cur][(wm * 64 + t * 16 + l16) * LDS_STRIDE];
            v8h lo = *(const v8h*)(ar + half * 16);
            v8h hi = *(const v8h*)(ar + half * 16 + 8);
            v16h afrag = __builtin_shufflevector(lo, hi,
                          0,1,2,3,4,5,6,7,8,9,10,11,12,13,14,15);
            #pragma unroll
            for (int u = 0; u < 4; ++u)
                acc[t][u] = __builtin_amdgcn_wmma_f32_16x16x32_f16(
                    /*neg_a=*/false, afrag,
                    /*neg_b=*/false, bfrag[u],
                    /*c_mod=*/(short)0, acc[t][u],
                    /*reuse_a=*/false, /*reuse_b=*/false);
        }

        // stage next slab into the other buffer (no conflict with 'cur' readers)
        if (kt + 1 < nk) {
            #pragma unroll
            for (int i = 0; i < 4; ++i) {
                const int r = i * 64 + rs;
                _Float16* pa = &As[nxt][r * LDS_STRIDE + cs * 4];
                pa[0] = (_Float16)ax[i].x; pa[1] = (_Float16)ax[i].y;
                pa[2] = (_Float16)ax[i].z; pa[3] = (_Float16)ax[i].w;
                _Float16* pb = &Bs[nxt][r * LDS_STRIDE + cs * 4];
                pb[0] = (_Float16)bw[i].x; pb[1] = (_Float16)bw[i].y;
                pb[2] = (_Float16)bw[i].z; pb[3] = (_Float16)bw[i].w;
            }
        }
        __syncthreads();
    }

    // ---- epilogue: fuse per-channel scale + bias, write fp32 ----
    // C/D layout: VGPR r -> M = r + 8*half ; N = lane%16
    #pragma unroll
    for (int u = 0; u < 4; ++u) {
        const int col = nBase + wn * 64 + u * 16 + l16;
        const float s = scale[col];
        const float b = bias[col];
        #pragma unroll
        for (int t = 0; t < 4; ++t) {
            const int row0 = mBase + wm * 64 + t * 16 + half * 8;
            #pragma unroll
            for (int r = 0; r < 8; ++r) {
                out[(size_t)(row0 + r) * N + col] = acc[t][u][r] * s + b;
            }
        }
    }
}

extern "C" void kernel_launch(void* const* d_in, const int* in_sizes, int n_in,
                              void* d_out, int out_size, void* d_ws, size_t ws_size,
                              hipStream_t stream) {
    const float* x     = (const float*)d_in[0];
    const int*   w     = (const int*)  d_in[1];
    const float* scale = (const float*)d_in[2];
    const float* bias  = (const float*)d_in[3];
    float*       out   = (float*)d_out;

    const int OUT = in_sizes[2];             // 11008
    const int IN  = in_sizes[1] / OUT;       // 4096
    const int M   = in_sizes[0] / IN;        // 8192

    dim3 grid(M / TILE_M, OUT / TILE_N);     // 32 x 43
    qlinear_wmma_f16<<<grid, NTHREADS, 0, stream>>>(x, w, scale, bias, out, M, IN, OUT);
}
